// ActorNetSpiking_77970836291971
// MI455X (gfx1250) — matrix operand order repaired and checked by reading, not compiled, not run on hardware
//
#include <hip/hip_runtime.h>

typedef __attribute__((ext_vector_type(16))) _Float16 v16h;
typedef __attribute__((ext_vector_type(8)))  _Float16 v8h;
typedef __attribute__((ext_vector_type(8)))  float    v8f;
typedef __attribute__((ext_vector_type(4)))  float    v4f;
typedef __attribute__((ext_vector_type(4)))  unsigned int v4u;
typedef __attribute__((ext_vector_type(8)))  int      v8i;
typedef __attribute__((ext_vector_type(4)))  int      v4i;

#define BATCH  4096
#define STATE  1024
#define HIDDEN 4096
#define ACTION 512
#define TSTEPS 16

#define CDECAY 0.5f
#define VDECAY 0.75f
#define VTH    0.5f

#define NT1    (HIDDEN/16)   // 256 B n-tiles layer 1
#define NT2    (ACTION/16)   // 32  B n-tiles layer 2

#define BM1 128              // rows per workgroup (both layers)
#define KC  128              // k-chunk staged in LDS
#define TSTR (KC + 8)        // LDS tile stride (halves): 272B rows -> conflict-free b128

#if defined(__has_builtin)
#if __has_builtin(__builtin_amdgcn_tensor_load_to_lds) && \
    __has_builtin(__builtin_amdgcn_s_wait_tensorcnt)
#define HAVE_TDM 1
#endif
#endif

__device__ __forceinline__ v16h cat16(v8h lo, v8h hi) {
  return __builtin_shufflevector(lo, hi, 0,1,2,3,4,5,6,7,8,9,10,11,12,13,14,15);
}

#define WMMA_F16(af, bf, c) \
  __builtin_amdgcn_wmma_f32_16x16x32_f16(false, (af), false, (bf), (short)0, (c), false, false)

#ifdef HAVE_TDM
// Issue one TDM 2D tile load: BM1 rows x KC halves (f16), row stride tensor_w
// halves, into LDS at lds_off with 16B padding appended per 256B row
// (pad_interval code 5 = 64 dwords, pad_amount code 3 = 4 dwords -> TSTR rows).
__device__ __forceinline__ void tdm_load_tile(unsigned int lds_off,
                                              const _Float16* gptr,
                                              unsigned int tensor_w) {
  unsigned long long ga = (unsigned long long)gptr;
  v4u g0;
  g0[0] = 1u;                                   // count=1, user descriptor
  g0[1] = lds_off;                              // lds_addr (bytes)
  g0[2] = (unsigned int)ga;                     // global_addr[31:0]
  g0[3] = (unsigned int)(ga >> 32) | (2u << 30);// global_addr[56:32] | type=2
  v8i g1;
  g1[0] = (int)((1u << 16) | (1u << 20) | (5u << 22) | (3u << 25));
  //        data_size=1(2B)  pad_enable   pad_interval=64dw pad_amount=4dw
  g1[1] = (int)((tensor_w & 0xFFFFu) << 16);    // tensor_dim0[15:0] @ bits63:48
  g1[2] = (int)((tensor_w >> 16) |              // tensor_dim0[31:16]
                ((unsigned)BM1 & 0xFFFFu) << 16);// tensor_dim1[15:0]
  g1[3] = (int)(((unsigned)KC) << 16);          // tensor_dim1[31:16]=0 | tile_dim0=KC
  g1[4] = (int)BM1;                             // tile_dim1=BM1 rows, tile_dim2=0
  g1[5] = (int)tensor_w;                        // tensor_dim0_stride[31:0] (halves)
  g1[6] = 0;                                    // stride[47:32]=0 | dim1_stride lo
  g1[7] = 0;
  v4i z4 = {0, 0, 0, 0};
  v8i z8 = {0, 0, 0, 0, 0, 0, 0, 0};
  __builtin_amdgcn_tensor_load_to_lds(g0, g1, z4, z4, z8, 0);  // 6-arg toolchain form
}
#endif

// ---------------- prep: x [B,S,T] f32 -> xT [T,B,S] f16 ----------------
__global__ void k_xpose(const float* __restrict__ x, _Float16* __restrict__ xT) {
  size_t idx = (size_t)blockIdx.x * blockDim.x + threadIdx.x;   // over B*S
  const v4f* px = (const v4f*)(x + idx * TSTEPS);
  v4f a = px[0], b = px[1], c = px[2], d = px[3];
  float f[16] = {a[0],a[1],a[2],a[3], b[0],b[1],b[2],b[3],
                 c[0],c[1],c[2],c[3], d[0],d[1],d[2],d[3]};
#pragma unroll
  for (int t = 0; t < TSTEPS; ++t)
    xT[(size_t)t * BATCH * STATE + idx] = (_Float16)f[t];       // coalesced per t
}

// ---- prep: W [N,K] f32 -> fragment-swizzled f16 B operands -------------
__global__ void k_wswz(const float* __restrict__ W, _Float16* __restrict__ Ws,
                       int NT, int K) {
  int gid  = blockIdx.x * blockDim.x + threadIdx.x;
  int lane = gid & 31;
  int frag = gid >> 5;                  // frag = kt*NT + nt
  int nt = frag % NT;
  int kt = frag / NT;
  int n = nt * 16 + (lane & 15);
  int k = kt * 32 + (lane >> 4) * 8;
  const float* src = W + (size_t)n * K + k;
  v4f s0 = *(const v4f*)(src +  0), s1 = *(const v4f*)(src +  4);
  v4f s2 = *(const v4f*)(src + 16), s3 = *(const v4f*)(src + 20);
  v16h d;
  d[ 0]=(_Float16)s0[0]; d[ 1]=(_Float16)s0[1]; d[ 2]=(_Float16)s0[2]; d[ 3]=(_Float16)s0[3];
  d[ 4]=(_Float16)s1[0]; d[ 5]=(_Float16)s1[1]; d[ 6]=(_Float16)s1[2]; d[ 7]=(_Float16)s1[3];
  d[ 8]=(_Float16)s2[0]; d[ 9]=(_Float16)s2[1]; d[10]=(_Float16)s2[2]; d[11]=(_Float16)s2[3];
  d[12]=(_Float16)s3[0]; d[13]=(_Float16)s3[1]; d[14]=(_Float16)s3[2]; d[15]=(_Float16)s3[3];
  *(v16h*)(Ws + ((size_t)frag * 32 + lane) * 16) = d;
}

// ------------- layer 1 step: u1 = 0.5u1 + x_t@W1^T + b1; LIF ------------
// grid (BATCH/128, HIDDEN/128), block 256 (8 waves: 4 wm x 2 wn)
__global__ __launch_bounds__(256, 1) void k_layer1(
    const _Float16* __restrict__ xT, const _Float16* __restrict__ W1s,
    const float* __restrict__ b1, float* __restrict__ u1ws,
    _Float16* __restrict__ v1ws, _Float16* __restrict__ s1ws,
    float* __restrict__ out1, int t) {
  __shared__ _Float16 atile[2 * BM1 * TSTR];   // double-buffered (~68 KB)

  const int tid  = threadIdx.x;
  const int wv   = tid >> 5, lane = tid & 31;
  const int lrow = lane & 15, lhi = lane >> 4;
  const int wm   = wv >> 1, wn = wv & 1;
  const int bm0  = blockIdx.x * BM1;
  const int n0   = blockIdx.y * 128 + wn * 64;   // wave covers 4 n-tiles
  const int m0   = wm * 32;                      // wave covers 2 m-tiles

  float b1v[4];
#pragma unroll
  for (int j = 0; j < 4; ++j) b1v[j] = b1[n0 + j * 16 + lrow];

  v8f acc[2][4];
#pragma unroll
  for (int mt = 0; mt < 2; ++mt)
#pragma unroll
    for (int j = 0; j < 4; ++j) {
#pragma unroll
      for (int e = 0; e < 8; ++e) {
        if (t > 0) {
          size_t idx = (size_t)(bm0 + m0 + mt * 16 + e + 8 * lhi) * HIDDEN
                       + (n0 + j * 16 + lrow);
          acc[mt][j][e] = u1ws[idx] * CDECAY + b1v[j];
        } else {
          acc[mt][j][e] = b1v[j];
        }
      }
    }

  const _Float16* xrow = xT + (size_t)t * BATCH * STATE;
#ifdef HAVE_TDM
  const unsigned int ldsA = (unsigned int)(unsigned long long)&atile[0];
  const unsigned int ldsB = (unsigned int)(unsigned long long)&atile[BM1 * TSTR];
  const _Float16* gtile = xrow + (size_t)bm0 * STATE;
  if (wv == 0) tdm_load_tile(ldsA, gtile, STATE);
#endif
  for (int kc = 0; kc < STATE / KC; ++kc) {
#ifdef HAVE_TDM
    const _Float16* curp = &atile[(kc & 1) ? BM1 * TSTR : 0];
    if (wv == 0) __builtin_amdgcn_s_wait_tensorcnt(0);  // current tile landed
    __syncthreads();
    if ((wv == 0) && (kc + 1 < STATE / KC))             // DMA next tile async
      tdm_load_tile((kc & 1) ? ldsA : ldsB, gtile + (size_t)(kc + 1) * KC, STATE);
#else
    const _Float16* curp = atile;
    __syncthreads();
    {
      const int r = tid >> 1, c = (tid & 1) * 64;
      const _Float16* src = xrow + (size_t)(bm0 + r) * STATE + kc * KC + c;
      _Float16* dst = &atile[r * TSTR + c];
#pragma unroll
      for (int i = 0; i < 8; ++i) *(v8h*)(dst + i * 8) = *(const v8h*)(src + i * 8);
    }
    __syncthreads();
#endif
    v16h bf[2][4];
    auto ldb = [&](int kt, v16h* arr) {
      const _Float16* bp = W1s + (((size_t)kt * NT1 + (n0 >> 4)) * 32 + lane) * 16;
#pragma unroll
      for (int j = 0; j < 4; ++j) arr[j] = *(const v16h*)(bp + (size_t)j * 512);
    };
    ldb(kc * 4, bf[0]);
#pragma unroll
    for (int ks = 0; ks < 4; ++ks) {
      if (ks < 3) ldb(kc * 4 + ks + 1, bf[(ks + 1) & 1]);  // double-buffer B
      const _Float16* ap = &curp[(m0 + lrow) * TSTR + ks * 32 + lhi * 8];
      v16h a0 = cat16(*(const v8h*)ap, *(const v8h*)(ap + 16));
      const _Float16* ap2 = ap + 16 * TSTR;
      v16h a1 = cat16(*(const v8h*)ap2, *(const v8h*)(ap2 + 16));
#pragma unroll
      for (int j = 0; j < 4; ++j) acc[0][j] = WMMA_F16(a0, bf[ks & 1][j], acc[0][j]);
#pragma unroll
      for (int j = 0; j < 4; ++j) acc[1][j] = WMMA_F16(a1, bf[ks & 1][j], acc[1][j]);
    }
  }

  // ---- LIF + state I/O (each element owned exclusively by this thread) ----
#pragma unroll
  for (int mt = 0; mt < 2; ++mt)
#pragma unroll
    for (int j = 0; j < 4; ++j)
#pragma unroll
      for (int e = 0; e < 8; ++e) {
        size_t idx = (size_t)(bm0 + m0 + mt * 16 + e + 8 * lhi) * HIDDEN
                     + (n0 + j * 16 + lrow);
        float un = acc[mt][j][e];
        float vo = (t > 0) ? (float)v1ws[idx] : 0.f;
        float vn = (vo > VTH) ? un : vo * VDECAY + un;   // v*decay*(1-s_old)+u
        _Float16 s = (vn > VTH) ? (_Float16)1.0f : (_Float16)0.0f;
        u1ws[idx] = un;
        v1ws[idx] = (_Float16)vn;
        s1ws[idx] = s;
        float so = (float)s;
        out1[idx] = (t > 0) ? out1[idx] + so : so;       // accumulate counts
      }
}

// ------------- layer 2 step: u2 = 0.5u2 + s1@W2^T + b2; LIF -------------
// grid (BATCH/128, ACTION/64), block 256 (8 waves: 4 wm x 2 wn)
__global__ __launch_bounds__(256, 1) void k_layer2(
    const _Float16* __restrict__ s1ws, const _Float16* __restrict__ W2s,
    const float* __restrict__ b2, float* __restrict__ u2ws,
    _Float16* __restrict__ v2ws, float* __restrict__ out2, int t) {
  __shared__ _Float16 atile[2 * BM1 * TSTR];   // double-buffered (~68 KB)

  const int tid  = threadIdx.x;
  const int wv   = tid >> 5, lane = tid & 31;
  const int lrow = lane & 15, lhi = lane >> 4;
  const int wm   = wv >> 1, wn = wv & 1;
  const int bm0  = blockIdx.x * BM1;
  const int n0   = blockIdx.y * 64 + wn * 32;    // wave covers 2 n-tiles
  const int m0   = wm * 32;                      // wave covers 2 m-tiles

  float b2v[2];
#pragma unroll
  for (int j = 0; j < 2; ++j) b2v[j] = b2[n0 + j * 16 + lrow];

  v8f acc[2][2];
#pragma unroll
  for (int mt = 0; mt < 2; ++mt)
#pragma unroll
    for (int j = 0; j < 2; ++j) {
#pragma unroll
      for (int e = 0; e < 8; ++e) {
        if (t > 0) {
          size_t idx = (size_t)(bm0 + m0 + mt * 16 + e + 8 * lhi) * ACTION
                       + (n0 + j * 16 + lrow);
          acc[mt][j][e] = u2ws[idx] * CDECAY + b2v[j];
        } else {
          acc[mt][j][e] = b2v[j];
        }
      }
    }

#ifdef HAVE_TDM
  const unsigned int ldsA = (unsigned int)(unsigned long long)&atile[0];
  const unsigned int ldsB = (unsigned int)(unsigned long long)&atile[BM1 * TSTR];
  const _Float16* gtile = s1ws + (size_t)bm0 * HIDDEN;
  if (wv == 0) tdm_load_tile(ldsA, gtile, HIDDEN);
#endif
  for (int kc = 0; kc < HIDDEN / KC; ++kc) {   // 32 chunks
#ifdef HAVE_TDM
    const _Float16* curp = &atile[(kc & 1) ? BM1 * TSTR : 0];
    if (wv == 0) __builtin_amdgcn_s_wait_tensorcnt(0);
    __syncthreads();
    if ((wv == 0) && (kc + 1 < HIDDEN / KC))
      tdm_load_tile((kc & 1) ? ldsA : ldsB, gtile + (size_t)(kc + 1) * KC, HIDDEN);
#else
    const _Float16* curp = atile;
    __syncthreads();
    {
      const int r = tid >> 1, c = (tid & 1) * 64;
      const _Float16* src = s1ws + (size_t)(bm0 + r) * HIDDEN + kc * KC + c;
      _Float16* dst = &atile[r * TSTR + c];
#pragma unroll
      for (int i = 0; i < 8; ++i) *(v8h*)(dst + i * 8) = *(const v8h*)(src + i * 8);
    }
    __syncthreads();
#endif
    v16h bf[2][2];
    auto ldb = [&](int kt, v16h* arr) {
      const _Float16* bp = W2s + (((size_t)kt * NT2 + (n0 >> 4)) * 32 + lane) * 16;
#pragma unroll
      for (int j = 0; j < 2; ++j) arr[j] = *(const v16h*)(bp + (size_t)j * 512);
    };
    ldb(kc * 4, bf[0]);
#pragma unroll
    for (int ks = 0; ks < 4; ++ks) {
      if (ks < 3) ldb(kc * 4 + ks + 1, bf[(ks + 1) & 1]);
      const _Float16* ap = &curp[(m0 + lrow) * TSTR + ks * 32 + lhi * 8];
      v16h a0 = cat16(*(const v8h*)ap, *(const v8h*)(ap + 16));
      const _Float16* ap2 = ap + 16 * TSTR;
      v16h a1 = cat16(*(const v8h*)ap2, *(const v8h*)(ap2 + 16));
#pragma unroll
      for (int j = 0; j < 2; ++j) acc[0][j] = WMMA_F16(a0, bf[ks & 1][j], acc[0][j]);
#pragma unroll
      for (int j = 0; j < 2; ++j) acc[1][j] = WMMA_F16(a1, bf[ks & 1][j], acc[1][j]);
    }
  }

#pragma unroll
  for (int mt = 0; mt < 2; ++mt)
#pragma unroll
    for (int j = 0; j < 2; ++j)
#pragma unroll
      for (int e = 0; e < 8; ++e) {
        size_t idx = (size_t)(bm0 + m0 + mt * 16 + e + 8 * lhi) * ACTION
                     + (n0 + j * 16 + lrow);
        float un = acc[mt][j][e];
        float vo = (t > 0) ? (float)v2ws[idx] : 0.f;
        float vn = (vo > VTH) ? un : vo * VDECAY + un;
        _Float16 s = (vn > VTH) ? (_Float16)1.0f : (_Float16)0.0f;
        u2ws[idx] = un;
        v2ws[idx] = (_Float16)vn;
        float so = (float)s;
        out2[idx] = (t > 0) ? out2[idx] + so : so;
      }
}

// ---------------- final: scale counts by 1/T in place ------------------
__global__ void k_scale(float* __restrict__ p, size_t n4) {
  size_t i = ((size_t)blockIdx.x * blockDim.x + threadIdx.x) * 4;
  if (i < n4 * 4) {
    v4f v = *(v4f*)(p + i);
    v[0] *= (1.f / TSTEPS); v[1] *= (1.f / TSTEPS);
    v[2] *= (1.f / TSTEPS); v[3] *= (1.f / TSTEPS);
    *(v4f*)(p + i) = v;
  }
}

extern "C" void kernel_launch(void* const* d_in, const int* in_sizes, int n_in,
                              void* d_out, int out_size, void* d_ws, size_t ws_size,
                              hipStream_t stream) {
  (void)in_sizes; (void)n_in; (void)out_size; (void)ws_size;
  const float* x  = (const float*)d_in[0];
  const float* W1 = (const float*)d_in[1];
  const float* b1 = (const float*)d_in[2];
  const float* W2 = (const float*)d_in[3];
  const float* b2 = (const float*)d_in[4];

  float* out1 = (float*)d_out;
  float* out2 = out1 + (size_t)BATCH * HIDDEN;

  // workspace layout (~286 MB)
  _Float16* xT   = (_Float16*)d_ws;                         // 134 MB
  _Float16* W1s  = xT + (size_t)TSTEPS * BATCH * STATE;     // 8 MB
  _Float16* W2s  = W1s + (size_t)STATE * HIDDEN;            // 4 MB
  _Float16* v1ws = W2s + (size_t)HIDDEN * ACTION;           // 32 MB
  _Float16* s1ws = v1ws + (size_t)BATCH * HIDDEN;           // 32 MB
  _Float16* v2ws = s1ws + (size_t)BATCH * HIDDEN;           // 4 MB
  float*    u1ws = (float*)(v2ws + (size_t)BATCH * ACTION); // 64 MB
  float*    u2ws = u1ws + (size_t)BATCH * HIDDEN;           // 8 MB

  k_xpose<<<(BATCH * STATE) / 256, 256, 0, stream>>>(x, xT);
  k_wswz<<<((STATE / 32) * (HIDDEN / 16) * 32) / 256, 256, 0, stream>>>(W1, W1s, HIDDEN / 16, STATE);
  k_wswz<<<((HIDDEN / 32) * (ACTION / 16) * 32) / 256, 256, 0, stream>>>(W2, W2s, ACTION / 16, HIDDEN);

  dim3 g1(BATCH / BM1, HIDDEN / 128);
  dim3 g2(BATCH / BM1, ACTION / 64);
  for (int t = 0; t < TSTEPS; ++t) {
    k_layer1<<<g1, 256, 0, stream>>>(xT, W1s, b1, u1ws, v1ws, s1ws, out1, t);
    k_layer2<<<g2, 256, 0, stream>>>(s1ws, W2s, b2, u2ws, v2ws, out2, t);
  }
  size_t total4 = ((size_t)BATCH * HIDDEN + (size_t)BATCH * ACTION) / 4;
  k_scale<<<(unsigned)((total4 + 255) / 256), 256, 0, stream>>>(out1, total4);
}
